// DeltaNet_31877247271434
// MI455X (gfx1250) — compile-verified, matrix operand order checked by reading, not run on hardware
//
#include <hip/hip_runtime.h>
#include <cstdint>

typedef __attribute__((ext_vector_type(16))) __bf16 v16bf;
typedef __attribute__((ext_vector_type(8)))  float  v8f;

#define Bc 2
#define Lc 4096
#define Dc 1024
#define Hc 4
#define DKc 256
#define Mc 8192            // B*L

#define WMMA_BF16(a, b, c) \
  __builtin_amdgcn_wmma_f32_16x16x32_bf16(false, (a), false, (b), (short)0, (c), false, false)

// fast silu/sigmoid via v_rcp (avoids the IEEE divide fixup chain)
__device__ __forceinline__ float silu_f(float x) {
  return x * __builtin_amdgcn_rcpf(1.f + __expf(-x));
}
__device__ __forceinline__ float sigmoid_f(float x) {
  return __builtin_amdgcn_rcpf(1.f + __expf(-x));
}

// Generic->LDS byte offset: LDS generic addresses carry the LDS offset in the
// low 32 bits.
__device__ __forceinline__ unsigned lds_off(const void* p) {
  return (unsigned)(unsigned long long)p;
}
// CDNA5 async copy: 16 bytes global -> LDS per active lane, tracked by ASYNCcnt.
__device__ __forceinline__ void async_copy16(unsigned ldsoff, const void* gptr) {
  asm volatile("global_load_async_to_lds_b128 %0, %1, off"
               :: "v"(ldsoff), "v"((unsigned long long)(uintptr_t)gptr)
               : "memory");
}
// ASYNCcnt decrements in issue order: waiting to <=2 releases the older tile
// while the newer (double-buffered) tile stays in flight.
__device__ __forceinline__ void wait_async_le2() {
  asm volatile("s_wait_asynccnt 2" ::: "memory");
}
__device__ __forceinline__ void wait_async0() {
  asm volatile("s_wait_asynccnt 0" ::: "memory");
}

// Load a 16x32 (MxK) A-style fragment (also used for B = Xᵀ when X stored row-major):
// lane L<16 -> row (row0+L), K[0..15]; lane L>=16 -> row (row0+L-16), K[16..31].
__device__ __forceinline__ v16bf ld_frag(const __bf16* base, int row0, int stride,
                                         int koff, int lane) {
  return *(const v16bf*)(base + (size_t)(row0 + (lane & 15)) * stride + koff +
                         ((lane >> 4) << 4));
}

// ---------------------------------------------------------------------------
// f32 -> bf16 elementwise convert (4 per thread)
// ---------------------------------------------------------------------------
__global__ __launch_bounds__(256) void cvt_bf16_kernel(const float* __restrict__ x,
                                                       __bf16* __restrict__ y) {
  int i = (blockIdx.x * 256 + threadIdx.x) * 4;
  float4 f = *(const float4*)(x + i);
  y[i + 0] = (__bf16)f.x; y[i + 1] = (__bf16)f.y;
  y[i + 2] = (__bf16)f.z; y[i + 3] = (__bf16)f.w;
}

// ---------------------------------------------------------------------------
// Wt[n*K+k] = bf16(W[k*N+n])  (32x32 LDS-tiled transpose)
// ---------------------------------------------------------------------------
__global__ __launch_bounds__(256) void transpose_cvt_kernel(
    const float* __restrict__ W, __bf16* __restrict__ Wt, int K, int N) {
  __shared__ float t[32][33];
  int kb = blockIdx.x * 32, nb = blockIdx.y * 32;
  int tx = threadIdx.x & 31, ty = threadIdx.x >> 5;
  for (int i = 0; i < 32; i += 8)
    t[ty + i][tx] = W[(size_t)(kb + ty + i) * N + nb + tx];
  __syncthreads();
  for (int i = 0; i < 32; i += 8)
    Wt[(size_t)(nb + ty + i) * K + kb + tx] = (__bf16)t[tx][ty + i];
}

// ---------------------------------------------------------------------------
// WMMA GEMM: C[M,N] = act(A[M,K] @ W[K,N] + bias)
// A: bf16 row-major [M][K]; Wt: bf16 pre-transposed [N][K].
// block = 256 threads (8 waves), tile 64x64, K-step 32.
// Tiles double-buffered via global_load_async_to_lds_b128: tile t+1 is issued
// before waiting on tile t (ASYNCcnt<=2), overlapping DMA with WMMA.
// ---------------------------------------------------------------------------
__global__ __launch_bounds__(256) void wmma_gemm_kernel(
    const __bf16* __restrict__ A, const __bf16* __restrict__ Wt,
    const float* __restrict__ bias, float* __restrict__ C,
    int M, int N, int K, int act) {
  __shared__ __attribute__((aligned(16))) __bf16 As[2][64 * 32];  // [m][k]
  __shared__ __attribute__((aligned(16))) __bf16 Bs[2][64 * 32];  // [n][k]
  const int tid = threadIdx.x, lane = tid & 31, wave = tid >> 5;
  const int m0 = blockIdx.x * 64, n0 = blockIdx.y * 64;
  const int wm = (wave & 3) * 16;
  const int wn = (wave >> 2) * 32;
  const int row = tid >> 2, seg = (tid & 3) * 8;  // 64 rows x 4 segs of 8 bf16
  const unsigned asd[2] = {lds_off(&As[0][row * 32 + seg]),
                           lds_off(&As[1][row * 32 + seg])};
  const unsigned bsd[2] = {lds_off(&Bs[0][row * 32 + seg]),
                           lds_off(&Bs[1][row * 32 + seg])};
  const __bf16* gA = A + (size_t)(m0 + row) * K + seg;
  const __bf16* gB = Wt + (size_t)(n0 + row) * K + seg;
  const int T = K >> 5;
  v8f acc0 = {0, 0, 0, 0, 0, 0, 0, 0};
  v8f acc1 = {0, 0, 0, 0, 0, 0, 0, 0};
  async_copy16(asd[0], gA);
  async_copy16(bsd[0], gB);
  for (int t = 0; t < T; ++t) {
    const int cur = t & 1;
    if (t + 1 < T) {
      async_copy16(asd[cur ^ 1], gA + (t + 1) * 32);
      async_copy16(bsd[cur ^ 1], gB + (t + 1) * 32);
      wait_async_le2();            // tile t done; tile t+1 still in flight
    } else {
      wait_async0();
    }
    __syncthreads();
    v16bf a  = ld_frag(As[cur], wm, 32, 0, lane);
    v16bf b0 = ld_frag(Bs[cur], wn, 32, 0, lane);
    v16bf b1 = ld_frag(Bs[cur], wn + 16, 32, 0, lane);
    acc0 = WMMA_BF16(a, b0, acc0);
    acc1 = WMMA_BF16(a, b1, acc1);
    __syncthreads();               // buffer `cur` free for tile t+2's issue
  }
  const int rr = (lane >> 4) * 8, cc = lane & 15;
#pragma unroll
  for (int j = 0; j < 8; ++j) {
    int orow = m0 + wm + rr + j;
    int c0 = n0 + wn + cc;
    float x0 = acc0[j], x1 = acc1[j];
    if (bias) { x0 += bias[c0]; x1 += bias[c0 + 16]; }
    if (act)  { x0 = silu_f(x0); x1 = silu_f(x1); }
    C[(size_t)orow * N + c0] = x0;
    C[(size_t)orow * N + c0 + 16] = x1;
  }
}

// ---------------------------------------------------------------------------
// Depthwise causal conv over (B,L,C=1024), optional SiLU
// ---------------------------------------------------------------------------
__global__ __launch_bounds__(256) void dwconv_kernel(
    const float* __restrict__ x, const float* __restrict__ w,
    float* __restrict__ y, int Kw, int act) {
  int idx = blockIdx.x * 256 + threadIdx.x;  // [0, M*D)
  int c = idx & (Dc - 1);
  int m = idx >> 10;
  int l = m & (Lc - 1);
  float acc = 0.f;
  for (int t = 0; t < Kw; ++t) {
    int ls = l - (Kw - 1) + t;
    if (ls >= 0) acc += x[(size_t)(m - l + ls) * Dc + c] * w[c * Kw + t];
  }
  if (act) acc = silu_f(acc);
  y[idx] = acc;
}

// ---------------------------------------------------------------------------
// beta = sigmoid(hs @ Wb), Wb: (1024,4). One block per row m.
// ---------------------------------------------------------------------------
__global__ __launch_bounds__(256) void beta_kernel(
    const float* __restrict__ hs, const float* __restrict__ Wb,
    float* __restrict__ beta) {
  __shared__ float red[4 * 256];
  int m = blockIdx.x, tid = threadIdx.x;
  float s0 = 0, s1 = 0, s2 = 0, s3 = 0;
  for (int d = tid; d < Dc; d += 256) {
    float x = hs[(size_t)m * Dc + d];
    s0 += x * Wb[d * 4 + 0]; s1 += x * Wb[d * 4 + 1];
    s2 += x * Wb[d * 4 + 2]; s3 += x * Wb[d * 4 + 3];
  }
  red[0 * 256 + tid] = s0; red[1 * 256 + tid] = s1;
  red[2 * 256 + tid] = s2; red[3 * 256 + tid] = s3;
  __syncthreads();
  for (int off = 128; off > 0; off >>= 1) {
    if (tid < off)
      for (int h = 0; h < 4; ++h) red[h * 256 + tid] += red[h * 256 + tid + off];
    __syncthreads();
  }
  if (tid < 4) beta[m * 4 + tid] = sigmoid_f(red[tid * 256]);
}

// ---------------------------------------------------------------------------
// Router tail: logits = h1 @ r_W2 + r_b2 ; per-head softmax over 4
// ---------------------------------------------------------------------------
__global__ __launch_bounds__(256) void router2_kernel(
    const float* __restrict__ h1, const float* __restrict__ W2,
    const float* __restrict__ b2, float* __restrict__ rw) {
  __shared__ float red[256];
  __shared__ float lg[16];
  int m = blockIdx.x, tid = threadIdx.x;
  int o = tid & 15, p = tid >> 4;
  float s = 0.f;
  const float* hr = h1 + (size_t)m * 2048 + p * 128;
  for (int kk = 0; kk < 128; ++kk) s += hr[kk] * W2[(p * 128 + kk) * 16 + o];
  red[p * 16 + o] = s;
  __syncthreads();
  if (tid < 16) {
    float t = b2[tid];
    for (int pp = 0; pp < 16; ++pp) t += red[pp * 16 + tid];
    lg[tid] = t;
  }
  __syncthreads();
  if (tid < 16) {
    int g = (tid >> 2) << 2;
    float mx = fmaxf(fmaxf(lg[g], lg[g + 1]), fmaxf(lg[g + 2], lg[g + 3]));
    float den = __expf(lg[g] - mx) + __expf(lg[g + 1] - mx) +
                __expf(lg[g + 2] - mx) + __expf(lg[g + 3] - mx);
    rw[m * 16 + tid] = __expf(lg[tid] - mx) * __builtin_amdgcn_rcpf(den);
  }
}

// ---------------------------------------------------------------------------
// Mix 4 branches with router weights, RMSNorm over DV=256 -> bf16 output
// (feeds the final WMMA GEMM). Block per (m,h).
// ---------------------------------------------------------------------------
__global__ __launch_bounds__(256) void mix_norm_kernel(
    const float* __restrict__ lo, const float* __restrict__ mi,
    const float* __restrict__ de, const float* __restrict__ vv,
    const float* __restrict__ rw, const float* __restrict__ nw,
    __bf16* __restrict__ om) {
  __shared__ float red[256];
  int mh = blockIdx.x;
  int m = mh >> 2, h = mh & 3, d = threadIdx.x;
  size_t base = (size_t)m * Dc + h * 256 + d;
  const float* r = rw + m * 16 + h * 4;
  float o = r[0] * lo[base] + r[1] * mi[base] + r[2] * de[base] + r[3] * vv[base];
  red[d] = o * o;
  __syncthreads();
  for (int off = 128; off > 0; off >>= 1) {
    if (d < off) red[d] += red[d + off];
    __syncthreads();
  }
  float sc = rsqrtf(red[0] * (1.f / 256.f) + 1e-5f);
  om[base] = (__bf16)(o * sc * nw[d]);
}

// ---------------------------------------------------------------------------
// Chunkwise delta rule. Grid = B*H*4 blocks: each owns a 64-wide DV slice of
// the 256x256 state S (fp32 master in LDS, bf16 shadow Sb for WMMA operands).
// Sequential over 128 chunks of 32. LDS map in comments below (194688 B).
// ---------------------------------------------------------------------------
#define DELTA_SMEM 194688

__global__ __launch_bounds__(256) void delta_kernel(
    const float* __restrict__ qg, const float* __restrict__ kg,
    const float* __restrict__ vg, const float* __restrict__ betag,
    float* __restrict__ outg) {
  extern __shared__ char smem[];
  float*  S     = (float*)(smem);            // [256][64] f32
  __bf16* Sb    = (__bf16*)(smem + 65536);   // [64][256] bf16 (Sᵀ)
  __bf16* qi_b  = (__bf16*)(smem + 98304);   // [32][256]
  __bf16* ki_b  = (__bf16*)(smem + 114688);  // [32][256]
  __bf16* kT_b  = (__bf16*)(smem + 131072);  // [256][32]
  __bf16* w_b   = (__bf16*)(smem + 147456);  // [32][256]
  float*  vi    = (float*)(smem + 163840);   // [32][64]
  float*  u     = (float*)(smem + 172032);   // [32][64] (u, then u_i)
  __bf16* uib   = (__bf16*)(smem + 180224);  // [64][32] (u_iᵀ)
  float*  G     = (float*)(smem + 184320);   // [32][32] k·kᵀ (+ norm scratch)
  float*  Tm    = (float*)(smem + 188416);   // [32][32]
  __bf16* attnb = (__bf16*)(smem + 192512);  // [32][32]
  float*  betas = (float*)(smem + 194560);   // [32]

  const int tid = threadIdx.x, lane = tid & 31, wave = tid >> 5;
  const int blk = blockIdx.x;
  const int b = blk >> 4;
  const int h = (blk >> 2) & 3;
  const int dv0 = (blk & 3) * 64;
  const size_t mbase = (size_t)b * Lc;

  for (int i = tid; i < 256 * 64; i += 256) S[i] = 0.f;
  __syncthreads();

  for (int ch = 0; ch < Lc / 32; ++ch) {
    const int l0 = ch * 32;
    const int r = tid >> 3, s8 = tid & 7;
    // prefetch next chunk's rows while we compute this one
    if (ch + 1 < Lc / 32) {
      const size_t nrow = mbase + l0 + 32 + r;
      __builtin_prefetch(qg + nrow * Dc + h * 256 + s8 * 32, 0, 1);
      __builtin_prefetch(kg + nrow * Dc + h * 256 + s8 * 32, 0, 1);
      __builtin_prefetch(vg + nrow * Dc + h * 256 + dv0 + s8 * 8, 0, 1);
    }
    // ---- Sb = bf16(Sᵀ) snapshot of state at chunk start ----
    for (int i = tid; i < 256 * 64; i += 256) {
      int dk = i >> 6, dv = i & 63;
      Sb[dv * 256 + dk] = (__bf16)S[i];
    }
    {  // ---- q: load, l2norm, -> bf16 ----
      float fr[32]; float ss = 0.f;
      const float* p = qg + (mbase + l0 + r) * Dc + h * 256 + s8 * 32;
#pragma unroll
      for (int i = 0; i < 8; ++i) {
        float4 f = *(const float4*)(p + i * 4);
        fr[i * 4 + 0] = f.x; fr[i * 4 + 1] = f.y;
        fr[i * 4 + 2] = f.z; fr[i * 4 + 3] = f.w;
        ss += f.x * f.x + f.y * f.y + f.z * f.z + f.w * f.w;
      }
      G[r * 8 + s8] = ss;
      __syncthreads();
      float tot = 0.f;
#pragma unroll
      for (int i = 0; i < 8; ++i) tot += G[r * 8 + i];
      float rn = rsqrtf(tot + 1e-12f);
      __bf16* dst = qi_b + r * 256 + s8 * 32;
#pragma unroll
      for (int x = 0; x < 32; ++x) dst[x] = (__bf16)(fr[x] * rn);
      __syncthreads();
    }
    {  // ---- k: load, l2norm, -> bf16 + transposed copy ----
      float fr[32]; float ss = 0.f;
      const float* p = kg + (mbase + l0 + r) * Dc + h * 256 + s8 * 32;
#pragma unroll
      for (int i = 0; i < 8; ++i) {
        float4 f = *(const float4*)(p + i * 4);
        fr[i * 4 + 0] = f.x; fr[i * 4 + 1] = f.y;
        fr[i * 4 + 2] = f.z; fr[i * 4 + 3] = f.w;
        ss += f.x * f.x + f.y * f.y + f.z * f.z + f.w * f.w;
      }
      G[r * 8 + s8] = ss;
      __syncthreads();
      float tot = 0.f;
#pragma unroll
      for (int i = 0; i < 8; ++i) tot += G[r * 8 + i];
      float rn = rsqrtf(tot + 1e-12f);
      __bf16* dst = ki_b + r * 256 + s8 * 32;
#pragma unroll
      for (int x = 0; x < 32; ++x) {
        __bf16 kv = (__bf16)(fr[x] * rn);
        dst[x] = kv;
        kT_b[(s8 * 32 + x) * 32 + r] = kv;
      }
      __syncthreads();
    }
    {  // ---- v slice (fp32) + beta (already sigmoided by beta_kernel) ----
      const float* p = vg + (mbase + l0 + r) * Dc + h * 256 + dv0 + s8 * 8;
      float4 f0 = *(const float4*)p;
      float4 f1 = *(const float4*)(p + 4);
      float* dst = vi + r * 64 + s8 * 8;
      dst[0] = f0.x; dst[1] = f0.y; dst[2] = f0.z; dst[3] = f0.w;
      dst[4] = f1.x; dst[5] = f1.y; dst[6] = f1.z; dst[7] = f1.w;
      if (tid < 32) betas[tid] = betag[(mbase + l0 + tid) * 4 + h];
      __syncthreads();
    }
    {  // ---- WMMA: waves 0-3: G = k kᵀ ; waves 4-7: attn = tril(q kᵀ) ----
      const int wv = wave & 3;
      const int mi = (wv >> 1) * 16, ni = (wv & 1) * 16;
      const __bf16* Ab = (wave < 4) ? ki_b : qi_b;
      v8f t = {0, 0, 0, 0, 0, 0, 0, 0};
#pragma unroll
      for (int kk = 0; kk < 256; kk += 32) {
        v16bf a  = ld_frag(Ab, mi, 256, kk, lane);
        v16bf bb = ld_frag(ki_b, ni, 256, kk, lane);
        t = WMMA_BF16(a, bb, t);
      }
      const int rr = (lane >> 4) * 8, cc = lane & 15;
      if (wave < 4) {
#pragma unroll
        for (int j = 0; j < 8; ++j) G[(mi + rr + j) * 32 + ni + cc] = t[j];
      } else {
#pragma unroll
        for (int j = 0; j < 8; ++j) {
          int R = mi + rr + j, Cc = ni + cc;
          attnb[R * 32 + Cc] = (__bf16)((Cc <= R) ? t[j] : 0.f);
        }
      }
    }
    __syncthreads();
    // ---- T = (I + strict_lower(diag(beta)·G))⁻¹ by forward substitution ----
    if (wave == 0) {
      for (int i = 0; i < 32; ++i) {
        float a = (lane == i) ? 1.f : 0.f;
        float bi = betas[i];
        for (int j = 0; j < i; ++j) a -= bi * G[i * 32 + j] * Tm[j * 32 + lane];
        Tm[i * 32 + lane] = a;
      }
    }
    __syncthreads();
    {  // ---- u = T·(beta*v) fp32 ; w = T·(beta*k) -> bf16 ----
      const int i = tid >> 3;
      float au[8] = {0, 0, 0, 0, 0, 0, 0, 0};
      for (int j = 0; j < 32; ++j) {
        float tj = Tm[i * 32 + j] * betas[j];
        const float* vr = vi + j * 64 + s8 * 8;
#pragma unroll
        for (int x = 0; x < 8; ++x) au[x] += tj * vr[x];
      }
      float* du = u + i * 64 + s8 * 8;
#pragma unroll
      for (int x = 0; x < 8; ++x) du[x] = au[x];
      float aw[32];
#pragma unroll
      for (int x = 0; x < 32; ++x) aw[x] = 0.f;
      for (int j = 0; j < 32; ++j) {
        float tj = Tm[i * 32 + j] * betas[j];
        const __bf16* kr = ki_b + j * 256 + s8 * 32;
#pragma unroll
        for (int x = 0; x < 32; ++x) aw[x] += tj * (float)kr[x];
      }
      __bf16* dw = w_b + i * 256 + s8 * 32;
#pragma unroll
      for (int x = 0; x < 32; ++x) dw[x] = (__bf16)aw[x];
    }
    __syncthreads();
    const int mi = (wave & 1) * 16, ni = (wave >> 1) * 16;
    const int rr = (lane >> 4) * 8, cc = lane & 15;
    {  // ---- u_i = u - w@S (WMMA, one 16x16 tile per wave) ----
      v8f p = {0, 0, 0, 0, 0, 0, 0, 0};
#pragma unroll
      for (int kk = 0; kk < 256; kk += 32) {
        v16bf a  = ld_frag(w_b, mi, 256, kk, lane);
        v16bf bb = ld_frag(Sb, ni, 256, kk, lane);
        p = WMMA_BF16(a, bb, p);
      }
#pragma unroll
      for (int j = 0; j < 8; ++j) {
        int R = mi + rr + j, Cc = ni + cc;
        float val = u[R * 64 + Cc] - p[j];
        u[R * 64 + Cc] = val;
        uib[Cc * 32 + R] = (__bf16)val;
      }
    }
    __syncthreads();
    {  // ---- o = q@S + attn@u_i -> global ----
      v8f o = {0, 0, 0, 0, 0, 0, 0, 0};
#pragma unroll
      for (int kk = 0; kk < 256; kk += 32) {
        v16bf a  = ld_frag(qi_b, mi, 256, kk, lane);
        v16bf bb = ld_frag(Sb, ni, 256, kk, lane);
        o = WMMA_BF16(a, bb, o);
      }
      v16bf aa = ld_frag(attnb, mi, 32, 0, lane);
      v16bf bu = ld_frag(uib, ni, 32, 0, lane);
      o = WMMA_BF16(aa, bu, o);
#pragma unroll
      for (int j = 0; j < 8; ++j) {
        int R = mi + rr + j, Cc = ni + cc;
        outg[(mbase + l0 + R) * Dc + h * 256 + dv0 + Cc] = o[j];
      }
    }
    {  // ---- S += kᵀ @ u_i  (64 tiles, 8 per wave, C from/to fp32 LDS) ----
#pragma unroll
      for (int tt = 0; tt < 8; ++tt) {
        int id = wave * 8 + tt;
        int smi = (id >> 2) * 16;
        int sni = (id & 3) * 16;
        v8f c = {0, 0, 0, 0, 0, 0, 0, 0};
#pragma unroll
        for (int j = 0; j < 8; ++j) c[j] = S[(smi + rr + j) * 64 + sni + cc];
        v16bf ak = ld_frag(kT_b, smi, 32, 0, lane);
        v16bf bu = ld_frag(uib, sni, 32, 0, lane);
        c = WMMA_BF16(ak, bu, c);
#pragma unroll
        for (int j = 0; j < 8; ++j) S[(smi + rr + j) * 64 + sni + cc] = c[j];
      }
    }
    __syncthreads();
  }
}

// ---------------------------------------------------------------------------
extern "C" void kernel_launch(void* const* d_in, const int* in_sizes, int n_in,
                              void* d_out, int out_size, void* d_ws, size_t ws_size,
                              hipStream_t stream) {
  (void)in_sizes; (void)n_in; (void)out_size; (void)ws_size;
  const float* hs  = (const float*)d_in[0];
  const float* Wq  = (const float*)d_in[1];
  const float* Wk  = (const float*)d_in[2];
  const float* Wv  = (const float*)d_in[3];
  const float* cq  = (const float*)d_in[4];
  const float* ck  = (const float*)d_in[5];
  const float* cv  = (const float*)d_in[6];
  const float* Wb  = (const float*)d_in[7];
  const float* lw  = (const float*)d_in[8];
  const float* mw  = (const float*)d_in[9];
  const float* rW1 = (const float*)d_in[10];
  const float* rb1 = (const float*)d_in[11];
  const float* rW2 = (const float*)d_in[12];
  const float* rb2 = (const float*)d_in[13];
  const float* nw  = (const float*)d_in[14];
  const float* Wo  = (const float*)d_in[15];
  float* out = (float*)d_out;
  char* ws = (char*)d_ws;
  const size_t MB = 1ull << 20;
  // f32 scratch (aliased regions noted)
  float* qp   = (float*)(ws + 0 * MB);      // pre-conv q (freed after conv)
  float* kp   = (float*)(ws + 32 * MB);     // pre-conv k (freed after conv)
  float* vp   = (float*)(ws + 64 * MB);     // pre-conv v (freed after conv)
  float* q    = (float*)(ws + 96 * MB);
  float* k    = (float*)(ws + 128 * MB);
  float* v    = (float*)(ws + 160 * MB);
  float* mid  = (float*)(ws + 192 * MB);
  float* dout = (float*)(ws + 224 * MB);
  float* beta = (float*)(ws + 256 * MB);
  float* rw   = (float*)(ws + 256 * MB + 131072);
  float* h1   = (float*)(ws + 0 * MB);      // 64MB, aliases qp+kp (after convs)
  float* loc  = (float*)(ws + 64 * MB);     // aliases vp (after conv_v)
  // bf16 operands for WMMA GEMMs
  __bf16* hs_b  = (__bf16*)(ws + 257 * MB);  // 16MB
  __bf16* Wqt   = (__bf16*)(ws + 273 * MB);  // 2MB each
  __bf16* Wkt   = (__bf16*)(ws + 275 * MB);
  __bf16* Wvt   = (__bf16*)(ws + 277 * MB);
  __bf16* rW1t  = (__bf16*)(ws + 279 * MB);  // 4MB
  __bf16* Wot   = (__bf16*)(ws + 283 * MB);  // 2MB
  __bf16* omixb = (__bf16*)(ws + 285 * MB);  // 16MB

  dim3 blk(256);
  // --- convert activations / weights to bf16 (weights transposed to [N][K]) ---
  cvt_bf16_kernel<<<dim3(Mc * Dc / 1024), blk, 0, stream>>>(hs, hs_b);
  transpose_cvt_kernel<<<dim3(Dc / 32, Dc / 32), blk, 0, stream>>>(Wq, Wqt, Dc, Dc);
  transpose_cvt_kernel<<<dim3(Dc / 32, Dc / 32), blk, 0, stream>>>(Wk, Wkt, Dc, Dc);
  transpose_cvt_kernel<<<dim3(Dc / 32, Dc / 32), blk, 0, stream>>>(Wv, Wvt, Dc, Dc);
  transpose_cvt_kernel<<<dim3(Dc / 32, 2048 / 32), blk, 0, stream>>>(rW1, rW1t, Dc, 2048);
  transpose_cvt_kernel<<<dim3(Dc / 32, Dc / 32), blk, 0, stream>>>(Wo, Wot, Dc, Dc);

  dim3 gproj(Mc / 64, Dc / 64);
  wmma_gemm_kernel<<<gproj, blk, 0, stream>>>(hs_b, Wqt, nullptr, qp, Mc, Dc, Dc, 0);
  wmma_gemm_kernel<<<gproj, blk, 0, stream>>>(hs_b, Wkt, nullptr, kp, Mc, Dc, Dc, 0);
  wmma_gemm_kernel<<<gproj, blk, 0, stream>>>(hs_b, Wvt, nullptr, vp, Mc, Dc, Dc, 0);

  dim3 gconv((Mc * Dc) / 256);
  dwconv_kernel<<<gconv, blk, 0, stream>>>(qp, cq, q, 4, 1);
  dwconv_kernel<<<gconv, blk, 0, stream>>>(kp, ck, k, 4, 1);
  dwconv_kernel<<<gconv, blk, 0, stream>>>(vp, cv, v, 4, 1);

  beta_kernel<<<dim3(Mc), blk, 0, stream>>>(hs, Wb, beta);

  wmma_gemm_kernel<<<dim3(Mc / 64, 2048 / 64), blk, 0, stream>>>(hs_b, rW1t, rb1, h1,
                                                                 Mc, 2048, Dc, 1);
  router2_kernel<<<dim3(Mc), blk, 0, stream>>>(h1, rW2, rb2, rw);

  dwconv_kernel<<<gconv, blk, 0, stream>>>(v, lw, loc, 7, 0);
  dwconv_kernel<<<gconv, blk, 0, stream>>>(v, mw, mid, 31, 0);

  (void)hipFuncSetAttribute((const void*)delta_kernel,
                            hipFuncAttributeMaxDynamicSharedMemorySize, DELTA_SMEM);
  delta_kernel<<<dim3(32), blk, DELTA_SMEM, stream>>>(q, k, v, beta, dout);

  mix_norm_kernel<<<dim3(Mc * 4), blk, 0, stream>>>(loc, mid, dout, v, rw, nw, omixb);

  wmma_gemm_kernel<<<dim3(Mc / 64, Dc / 64), blk, 0, stream>>>(omixb, Wot, nullptr, out,
                                                               Mc, Dc, Dc, 0);
}
